// MeTokenGNN_27453430956546
// MI455X (gfx1250) — matrix operand order repaired and checked by reading, not compiled, not run on hardware
//
#include <hip/hip_runtime.h>
#include <math.h>

// ---------------------------------------------------------------------------
// MeToken GNN layer for MI455X (gfx1250, wave32, WMMA).
// Edge MLP (105 GFLOP) + node FFN (13 GFLOP) run on v_wmma_f32_16x16x32_bf16.
// h_V (25.6MB) is L2-resident; edge messages are never materialized (fused
// scatter via global_atomic_add_f32). Weights pre-packed into WMMA B-fragment
// lane layout so each fragment is two b128 loads.
// ---------------------------------------------------------------------------

typedef float v8f __attribute__((ext_vector_type(8)));
typedef __bf16 v16bf __attribute__((ext_vector_type(16)));

struct Frag {
  union {
    v16bf v;
    uint4 q[2];
    unsigned short u[16];
  };
};

__device__ __forceinline__ unsigned short f2bf(float f) {
  union { float f; unsigned u; } c; c.f = f;
  unsigned u = c.u;
  u += 0x7FFFu + ((u >> 16) & 1u);           // round-to-nearest-even
  return (unsigned short)(u >> 16);
}

__device__ __forceinline__ float silu_f(float x) {
  return x / (1.0f + __expf(-x));
}

__device__ __forceinline__ v8f vzero() {
  v8f c;
#pragma unroll
  for (int i = 0; i < 8; ++i) c[i] = 0.0f;
  return c;
}

// A-fragment (16x32 bf16, MxK): lane = row (L%16), half=L/16.
// elements 0..7  -> K = kb*32 + half*8 + i
// elements 8..15 -> K = kb*32 + 16 + half*8 + (i-8)
// => two contiguous 16B chunks per lane.
__device__ __forceinline__ void loadA(Frag& a, const unsigned short* Lrow,
                                      int kb, int half) {
  const unsigned short* p = Lrow + kb * 32 + half * 8;
  a.q[0] = *(const uint4*)(p);
  a.q[1] = *(const uint4*)(p + 16);
}

// B-fragment pre-packed in workspace: frag-major [frag][lane][16 bf16].
__device__ __forceinline__ void loadB(Frag& b, const unsigned short* wf,
                                      int frag, int lane) {
  const uint4* p = (const uint4*)(wf + (size_t)frag * 512 + lane * 16);
  b.q[0] = p[0];
  b.q[1] = p[1];
}

__device__ __forceinline__ v8f wmma_bf16(const Frag& a, const Frag& b, v8f c) {
  return __builtin_amdgcn_wmma_f32_16x16x32_bf16(false, a.v, false, b.v,
                                                 (short)0, c, false, false);
}

// ---------------------------------------------------------------------------
// Utility kernels
// ---------------------------------------------------------------------------
__global__ void zero_kernel(float* p, long n) {
  long i = (long)blockIdx.x * 256 + threadIdx.x;
  long st = (long)gridDim.x * 256;
  for (; i < n; i += st) p[i] = 0.0f;
}

// Pack row-major fp32 W[K][Nn] -> bf16 B-fragment layout.
// frag = kb*NB + nb ; within frag: lane (0..31), elem i (0..15)
// k = kb*32 + (i&7) + (lane>=16)*8 + (i>=8)*16 ; col = nb*16 + (lane&15)
__global__ void pack_kernel(const float* __restrict__ W,
                            unsigned short* __restrict__ out, int K, int Nn) {
  int NB = Nn >> 4;
  long total = (long)K * Nn;
  for (long id = (long)blockIdx.x * 256 + threadIdx.x; id < total;
       id += (long)gridDim.x * 256) {
    long frag = id >> 9;
    int rem = (int)(id & 511);
    int lane = rem >> 4;
    int i = rem & 15;
    int kb = (int)(frag / NB), nb = (int)(frag % NB);
    int k = kb * 32 + (i & 7) + (lane >> 4) * 8 + (i >> 3) * 16;
    int col = nb * 16 + (lane & 15);
    out[id] = f2bf(W[(long)k * Nn + col]);
  }
}

// ---------------------------------------------------------------------------
// Edge MLP: per 16-edge tile, gather -> [16x256] bf16 in LDS ->
// 3x WMMA GEMM (SiLU between) -> atomic scatter into s[src], cnt[src].
// ---------------------------------------------------------------------------
#define WPB 4

__global__ __launch_bounds__(128) void edge_mlp_kernel(
    const float* __restrict__ hV, const int* __restrict__ srcI,
    const int* __restrict__ dstI, const unsigned short* __restrict__ w1f,
    const unsigned short* __restrict__ w2f,
    const unsigned short* __restrict__ w3f, const float* __restrict__ bw1,
    const float* __restrict__ bw2, const float* __restrict__ bw3,
    float* __restrict__ sOut, float* __restrict__ cnt, int nTiles) {
  __shared__ __align__(16) unsigned short ldsIn[WPB][16 * 256];
  __shared__ __align__(16) unsigned short ldsMid[WPB][16 * 128];

  const int wave = threadIdx.x >> 5;
  const int lane = threadIdx.x & 31;
  const int half = lane >> 4;
  const int lr = lane & 15;

  for (int tile = blockIdx.x * WPB + wave; tile < nTiles;
       tile += gridDim.x * WPB) {
    const long base = (long)tile * 16;
    unsigned short* L = ldsIn[wave];
    unsigned short* M = ldsMid[wave];

    // ---- gather [16 edges x 256] as bf16 (src||dst concat), float4 loads
    for (int it = 0; it < 32; ++it) {
      int slot = it * 32 + lane;            // 0..1023 (16 rows x 64 vec4)
      int row = slot >> 6;
      int col = (slot & 63) * 4;
      long e = base + row;
      const float* p = (col < 128)
                           ? (hV + (long)srcI[e] * 128 + col)
                           : (hV + (long)dstI[e] * 128 + (col - 128));
      float4 v = *(const float4*)p;
      *(unsigned int*)&L[row * 256 + col] =
          (unsigned)f2bf(v.x) | ((unsigned)f2bf(v.y) << 16);
      *(unsigned int*)&L[row * 256 + col + 2] =
          (unsigned)f2bf(v.z) | ((unsigned)f2bf(v.w) << 16);
    }
    asm volatile("" ::: "memory");

    // ---- layer 1: [16x256] @ [256x128]
    Frag A1[8];
#pragma unroll
    for (int kb = 0; kb < 8; ++kb) loadA(A1[kb], L + lr * 256, kb, half);
#pragma unroll
    for (int nb = 0; nb < 8; ++nb) {
      v8f acc = vzero();
#pragma unroll
      for (int kb = 0; kb < 8; ++kb) {
        Frag B;
        loadB(B, w1f, kb * 8 + nb, lane);
        acc = wmma_bf16(A1[kb], B, acc);
      }
      float bias = bw1[nb * 16 + lr];
#pragma unroll
      for (int r = 0; r < 8; ++r)
        M[(r + half * 8) * 128 + nb * 16 + lr] = f2bf(silu_f(acc[r] + bias));
    }
    asm volatile("" ::: "memory");

    // ---- layer 2: [16x128] @ [128x128]
    Frag A2[4];
#pragma unroll
    for (int kb = 0; kb < 4; ++kb) loadA(A2[kb], M + lr * 128, kb, half);
    v8f C2[8];
#pragma unroll
    for (int nb = 0; nb < 8; ++nb) {
      v8f acc = vzero();
#pragma unroll
      for (int kb = 0; kb < 4; ++kb) {
        Frag B;
        loadB(B, w2f, kb * 8 + nb, lane);
        acc = wmma_bf16(A2[kb], B, acc);
      }
      C2[nb] = acc;
    }
#pragma unroll
    for (int nb = 0; nb < 8; ++nb) {
      float bias = bw2[nb * 16 + lr];
#pragma unroll
      for (int r = 0; r < 8; ++r)
        M[(r + half * 8) * 128 + nb * 16 + lr] = f2bf(silu_f(C2[nb][r] + bias));
    }
    asm volatile("" ::: "memory");

    // ---- layer 3: [16x128] @ [128x128], then atomic scatter
    Frag A3[4];
#pragma unroll
    for (int kb = 0; kb < 4; ++kb) loadA(A3[kb], M + lr * 128, kb, half);

    int nodes[8];
#pragma unroll
    for (int r = 0; r < 8; ++r) nodes[r] = srcI[base + r + half * 8];

#pragma unroll
    for (int nb = 0; nb < 8; ++nb) {
      v8f acc = vzero();
#pragma unroll
      for (int kb = 0; kb < 4; ++kb) {
        Frag B;
        loadB(B, w3f, kb * 8 + nb, lane);
        acc = wmma_bf16(A3[kb], B, acc);
      }
      float bias = bw3[nb * 16 + lr];
#pragma unroll
      for (int r = 0; r < 8; ++r)
        atomicAdd(&sOut[(long)nodes[r] * 128 + nb * 16 + lr], acc[r] + bias);
    }
    if (lane < 16) atomicAdd(&cnt[srcI[base + lane]], 1.0f);
  }
}

// ---------------------------------------------------------------------------
// x = h_V + s/max(cnt,1) (in place in s); accumulate BN0 sums per feature.
// ---------------------------------------------------------------------------
__global__ __launch_bounds__(128) void combine_stats_kernel(
    const float* __restrict__ hV, float* __restrict__ x,
    const float* __restrict__ cnt, float* __restrict__ sum,
    float* __restrict__ sq, int N) {
  int f = threadIdx.x;  // 128 features
  float s = 0.0f, s2 = 0.0f;
  for (int n = blockIdx.x; n < N; n += gridDim.x) {
    float c = cnt[n];
    c = (c < 1.0f) ? 1.0f : c;
    float v = hV[(size_t)n * 128 + f] + x[(size_t)n * 128 + f] / c;
    x[(size_t)n * 128 + f] = v;
    s += v;
    s2 += v * v;
  }
  atomicAdd(&sum[f], s);
  atomicAdd(&sq[f], s2);
}

__global__ __launch_bounds__(128) void stats_kernel(const float* __restrict__ y,
                                                    float* __restrict__ sum,
                                                    float* __restrict__ sq,
                                                    int N) {
  int f = threadIdx.x;
  float s = 0.0f, s2 = 0.0f;
  for (int n = blockIdx.x; n < N; n += gridDim.x) {
    float v = y[(size_t)n * 128 + f];
    s += v;
    s2 += v * v;
  }
  atomicAdd(&sum[f], s);
  atomicAdd(&sq[f], s2);
}

__global__ __launch_bounds__(128) void finalize_bn_kernel(
    const float* __restrict__ sum, const float* __restrict__ sq,
    const float* __restrict__ g, const float* __restrict__ be,
    float* __restrict__ cA, float* __restrict__ cB, float invN) {
  int f = threadIdx.x;
  float m = sum[f] * invN;
  float var = sq[f] * invN - m * m;
  float r = rsqrtf(var + 1e-5f);
  float a = g[f] * r;
  cA[f] = a;
  cB[f] = be[f] - m * a;
}

// ---------------------------------------------------------------------------
// Node FFN: h = BN0(x); y = h + silu(h@d1+bd1)@d2+bd2. Per 16-node tile,
// the 512-wide intermediate is streamed through LDS in 128-col chunks.
// ---------------------------------------------------------------------------
__global__ __launch_bounds__(128) void node_ffn_kernel(
    const float* __restrict__ xbuf, const float* __restrict__ cA0,
    const float* __restrict__ cB0, const unsigned short* __restrict__ d1f,
    const unsigned short* __restrict__ d2f, const float* __restrict__ bd1,
    const float* __restrict__ bd2, float* __restrict__ y, int nTiles) {
  __shared__ __align__(16) unsigned short hL[WPB][16 * 128];
  __shared__ __align__(16) unsigned short tL[WPB][16 * 128];

  const int wave = threadIdx.x >> 5;
  const int lane = threadIdx.x & 31;
  const int half = lane >> 4;
  const int lr = lane & 15;

  for (int tile = blockIdx.x * WPB + wave; tile < nTiles;
       tile += gridDim.x * WPB) {
    const long base = (long)tile * 16;
    unsigned short* H = hL[wave];
    unsigned short* T = tL[wave];

    // stage normalized h into LDS (bf16)
    for (int it = 0; it < 16; ++it) {
      int slot = it * 32 + lane;  // 512 slots (16 rows x 32 vec4)
      int row = slot >> 5;
      int col = (slot & 31) * 4;
      float4 v = *(const float4*)(xbuf + (base + row) * 128 + col);
      float h0 = v.x * cA0[col + 0] + cB0[col + 0];
      float h1 = v.y * cA0[col + 1] + cB0[col + 1];
      float h2 = v.z * cA0[col + 2] + cB0[col + 2];
      float h3 = v.w * cA0[col + 3] + cB0[col + 3];
      *(unsigned int*)&H[row * 128 + col] =
          (unsigned)f2bf(h0) | ((unsigned)f2bf(h1) << 16);
      *(unsigned int*)&H[row * 128 + col + 2] =
          (unsigned)f2bf(h2) | ((unsigned)f2bf(h3) << 16);
    }
    asm volatile("" ::: "memory");

    Frag Ah[4];
#pragma unroll
    for (int kb = 0; kb < 4; ++kb) loadA(Ah[kb], H + lr * 128, kb, half);

    v8f C2[8];
#pragma unroll
    for (int nb = 0; nb < 8; ++nb) C2[nb] = vzero();

    for (int c = 0; c < 4; ++c) {
      // GEMM1 chunk: h[16x128] @ d1[:,c*128:(c+1)*128] -> silu -> T
#pragma unroll
      for (int j = 0; j < 8; ++j) {
        int nbg = c * 8 + j;  // d1 NB = 32
        v8f acc = vzero();
#pragma unroll
        for (int kb = 0; kb < 4; ++kb) {
          Frag B;
          loadB(B, d1f, kb * 32 + nbg, lane);
          acc = wmma_bf16(Ah[kb], B, acc);
        }
        float bias = bd1[nbg * 16 + lr];
#pragma unroll
        for (int r = 0; r < 8; ++r)
          T[(r + half * 8) * 128 + j * 16 + lr] = f2bf(silu_f(acc[r] + bias));
      }
      asm volatile("" ::: "memory");

      // GEMM2 partial: T[16x128] @ d2[c*128:(c+1)*128, :]
      Frag At[4];
#pragma unroll
      for (int k = 0; k < 4; ++k) loadA(At[k], T + lr * 128, k, half);
#pragma unroll
      for (int nb = 0; nb < 8; ++nb) {
#pragma unroll
        for (int k = 0; k < 4; ++k) {
          Frag B;
          loadB(B, d2f, (c * 4 + k) * 8 + nb, lane);  // d2 KB=16, NB=8
          C2[nb] = wmma_bf16(At[k], B, C2[nb]);
        }
      }
      asm volatile("" ::: "memory");
    }

    // epilogue: y = h + ffn
#pragma unroll
    for (int nb = 0; nb < 8; ++nb) {
      int col = nb * 16 + lr;
      float bias = bd2[col], a = cA0[col], b = cB0[col];
#pragma unroll
      for (int r = 0; r < 8; ++r) {
        long node = base + r + half * 8;
        float xv = xbuf[node * 128 + col];
        y[node * 128 + col] = xv * a + b + C2[nb][r] + bias;
      }
    }
  }
}

__global__ void bn_apply_kernel(float* __restrict__ y,
                                const float* __restrict__ cA,
                                const float* __restrict__ cB, long n) {
  long i = (long)blockIdx.x * 256 + threadIdx.x;
  long st = (long)gridDim.x * 256;
  for (; i < n; i += st) {
    int c = (int)(i & 127);
    y[i] = y[i] * cA[c] + cB[c];
  }
}

// ---------------------------------------------------------------------------
// Host launcher
// ---------------------------------------------------------------------------
extern "C" void kernel_launch(void* const* d_in, const int* in_sizes, int n_in,
                              void* d_out, int out_size, void* d_ws,
                              size_t ws_size, hipStream_t stream) {
  const float* hV = (const float*)d_in[0];
  const int* eidx = (const int*)d_in[1];
  // d_in[2] = batch_id (unused in forward)
  const float* w1 = (const float*)d_in[3];
  const float* bw1 = (const float*)d_in[4];
  const float* w2 = (const float*)d_in[5];
  const float* bw2 = (const float*)d_in[6];
  const float* w3 = (const float*)d_in[7];
  const float* bw3 = (const float*)d_in[8];
  const float* d1 = (const float*)d_in[9];
  const float* bd1 = (const float*)d_in[10];
  const float* d2 = (const float*)d_in[11];
  const float* bd2 = (const float*)d_in[12];
  const float* g0 = (const float*)d_in[13];
  const float* be0 = (const float*)d_in[14];
  const float* g1 = (const float*)d_in[15];
  const float* be1 = (const float*)d_in[16];

  const long N = (long)in_sizes[0] / 128;
  const long E = (long)in_sizes[1] / 2;
  const int* srcI = eidx;
  const int* dstI = eidx + E;

  auto al = [](size_t x) { return (x + 255) & ~(size_t)255; };
  char* ws = (char*)d_ws;
  size_t off = 0;
  float* sbuf = (float*)(ws + off);          off = al(off + (size_t)N * 128 * 4);
  float* cnt = (float*)(ws + off);           off = al(off + (size_t)N * 4);
  float* stats = (float*)(ws + off);
  size_t zeroEnd = off + 1024 * 4;           off = al(zeroEnd);
  unsigned short* w1f = (unsigned short*)(ws + off); off = al(off + 32768 * 2);
  unsigned short* w2f = (unsigned short*)(ws + off); off = al(off + 16384 * 2);
  unsigned short* w3f = (unsigned short*)(ws + off); off = al(off + 16384 * 2);
  unsigned short* d1f = (unsigned short*)(ws + off); off = al(off + 65536 * 2);
  unsigned short* d2f = (unsigned short*)(ws + off); off = al(off + 65536 * 2);

  float* sum0 = stats;       float* sq0 = stats + 128;
  float* cA0 = stats + 256;  float* cB0 = stats + 384;
  float* sum1 = stats + 512; float* sq1 = stats + 640;
  float* cA1 = stats + 768;  float* cB1 = stats + 896;

  float* yOut = (float*)d_out;

  // 1) zero accumulators (s, cnt, stats) — every call, graph-safe
  {
    long nz = (long)(zeroEnd / 4);
    int blocks = (int)((nz + 255) / 256);
    if (blocks > 16384) blocks = 16384;
    zero_kernel<<<blocks, 256, 0, stream>>>((float*)ws, nz);
  }

  // 2) pack weights to bf16 WMMA B-fragment layout
  auto packs = [&](const float* W, unsigned short* out, int K, int Nn) {
    long total = (long)K * Nn;
    int blocks = (int)((total + 255) / 256);
    pack_kernel<<<blocks, 256, 0, stream>>>(W, out, K, Nn);
  };
  packs(w1, w1f, 256, 128);
  packs(w2, w2f, 128, 128);
  packs(w3, w3f, 128, 128);
  packs(d1, d1f, 128, 512);
  packs(d2, d2f, 512, 128);

  // 3) edge MLP + fused scatter
  {
    int nTiles = (int)(E / 16);
    int blocks = (nTiles + WPB - 1) / WPB;
    edge_mlp_kernel<<<blocks, 128, 0, stream>>>(hV, srcI, dstI, w1f, w2f, w3f,
                                                bw1, bw2, bw3, sbuf, cnt,
                                                nTiles);
  }

  // 4) x = h_V + s/cnt ; BN0 stats
  combine_stats_kernel<<<512, 128, 0, stream>>>(hV, sbuf, cnt, sum0, sq0,
                                                (int)N);
  // 5) BN0 coefficients
  finalize_bn_kernel<<<1, 128, 0, stream>>>(sum0, sq0, g0, be0, cA0, cB0,
                                            1.0f / (float)N);
  // 6) FFN + residual -> y (in d_out)
  {
    int nTiles = (int)(N / 16);
    int blocks = (nTiles + WPB - 1) / WPB;
    node_ffn_kernel<<<blocks, 128, 0, stream>>>(sbuf, cA0, cB0, d1f, d2f, bd1,
                                                bd2, yOut, nTiles);
  }
  // 7) BN1 stats over y
  stats_kernel<<<512, 128, 0, stream>>>(yOut, sum1, sq1, (int)N);
  // 8) BN1 coefficients
  finalize_bn_kernel<<<1, 128, 0, stream>>>(sum1, sq1, g1, be1, cA1, cB1,
                                            1.0f / (float)N);
  // 9) normalize in place
  {
    long n = (long)N * 128;
    int blocks = (int)((n + 255) / 256);
    if (blocks > 16384) blocks = 16384;
    bn_apply_kernel<<<blocks, 256, 0, stream>>>(yOut, cA1, cB1, n);
  }
}
